// QNNAttentionModel_13615046328509
// MI455X (gfx1250) — matrix operand order repaired and checked by reading, not compiled, not run on hardware
//
#include <hip/hip_runtime.h>

// ---------------------------------------------------------------------------
// Types for CDNA5 WMMA (wave32): v_wmma_f32_16x16x32_bf16
// ---------------------------------------------------------------------------
typedef __attribute__((ext_vector_type(16))) __bf16 v16bf;
typedef __attribute__((ext_vector_type(8)))  __bf16 v8bf;
typedef __attribute__((ext_vector_type(8)))  float  v8f;

union AFrag { v16bf v; v8bf h[2]; };

// Hamilton product expansion tables.
// o_r = r@Wr - i@Wi - j@Wj - k@Wk
// o_i = r@Wi + i@Wr + j@Wk - k@Wj
// o_j = r@Wj - i@Wk + j@Wr + k@Wi
// o_k = r@Wk + i@Wj - j@Wi + k@Wr
// QCOMP[ci][co] = which W component, QSIGN[ci][co] = sign.
__device__ __constant__ int   QCOMP[4][4] = {{0,1,2,3},{1,0,3,2},{2,3,0,1},{3,2,1,0}};
__device__ __constant__ float QSIGN[4][4] = {{ 1.f, 1.f, 1.f, 1.f},
                                             {-1.f, 1.f,-1.f, 1.f},
                                             {-1.f, 1.f, 1.f,-1.f},
                                             {-1.f,-1.f, 1.f, 1.f}};

// ---------------------------------------------------------------------------
// Pack Hamilton-expanded recurrent weight into WMMA-B fragment layout (bf16).
// B fragment (32x16, 16-bit), this code's contiguous k order: lane L holds N = L&15, K = (L>=16?16:0)+e,
// e = 0..15 contiguous bf16 per lane (32 bytes -> 2x global_load_b128).
// Packed layout: [ntile][kt][lane][e], ntile over N/16=128, kt over K/32.
// K index k: rows 0..511 come from Wh (Hamilton state), rows 512..1023 (if
// Wx != nullptr) come from Wx (layer-1 input projection folded into GEMM).
// Row meaning: k' = ci*128 + hin (component-major h vector).
// Col meaning: n  = co*512 + gout, gout = gate*128 + hidx.
// ---------------------------------------------------------------------------
__global__ void pack_hamilton(const float* __restrict__ Wh,
                              const float* __restrict__ Wx,
                              int ktiles, __bf16* __restrict__ out) {
    size_t idx = (size_t)blockIdx.x * blockDim.x + threadIdx.x;
    size_t total = (size_t)128 * ktiles * 32 * 16;
    if (idx >= total) return;
    int e     = (int)(idx & 15);
    int lane  = (int)((idx >> 4) & 31);
    size_t rest = idx >> 9;
    int kt    = (int)(rest % (size_t)ktiles);
    int ntile = (int)(rest / (size_t)ktiles);

    int k = kt * 32 + (lane >> 4) * 16 + e;
    int n = ntile * 16 + (lane & 15);

    const float* W = Wh;
    int kk = k;
    if (kk >= 512) { W = Wx; kk -= 512; }
    int ci   = kk >> 7;
    int hin  = kk & 127;
    int co   = n >> 9;
    int gout = n & 511;
    float v = QSIGN[ci][co] * W[(size_t)QCOMP[ci][co] * 128 * 512 + (size_t)hin * 512 + gout];
    out[idx] = (__bf16)v;
}

// Pack Wp (512x128, rows indexed h*4+co from the reference reshape) into the
// packed-B layout with quaternion remap to our component-major h vector
// (k = co*128 + h). NT = 8, KT = 16.
__global__ void pack_wp(const float* __restrict__ Wp, __bf16* __restrict__ out) {
    size_t idx = (size_t)blockIdx.x * blockDim.x + threadIdx.x; // 65536 total
    int e     = (int)(idx & 15);
    int lane  = (int)((idx >> 4) & 31);
    size_t rest = idx >> 9;
    int kt    = (int)(rest & 15);
    int ntile = (int)(rest >> 4);
    int k = kt * 32 + (lane >> 4) * 16 + e;
    int n = ntile * 16 + (lane & 15);
    int row = (k & 127) * 4 + (k >> 7);      // h*4 + co
    out[idx] = (__bf16)Wp[(size_t)row * 128 + n];
}

__global__ void zero_f32(float* __restrict__ p, int n) {
    int i = blockIdx.x * blockDim.x + threadIdx.x;
    if (i < n) p[i] = 0.f;
}

// ---------------------------------------------------------------------------
// WMMA GEMM: C[M x N] = A[M x K] * Bpacked + (bias).
// A is bf16 row-major, possibly split in two 512-wide halves (A0 | A1) when
// K = 1024 (layer-1 recurrence with folded input projection).
// Block = 256 threads = 8 waves; each block owns one 16-row M tile and a
// 128-wide N panel; wave w computes the 16x16 tile at n = panel*128 + w*16.
// A tile is staged into LDS with GLOBAL_LOAD_ASYNC_TO_LDS_B128 (memory->LDS
// DMA path, tracked by ASYNCcnt) and re-read per the 16x32 A-fragment lane
// layout with ds_load_b128.
// ---------------------------------------------------------------------------
__global__ void __launch_bounds__(256)
wmma_gemm(const __bf16* __restrict__ A0,
          const __bf16* __restrict__ A1,
          const __bf16* __restrict__ Bp,
          float* __restrict__ C, int ldc, int ktiles,
          const float* __restrict__ bias) {
    __shared__ __bf16 As[16 * 1024];
    const int K    = ktiles * 32;
    const int mrow = blockIdx.x * 16;
    const int w    = threadIdx.x >> 5;
    const int lane = threadIdx.x & 31;
    const int ntile = blockIdx.y * 8 + w;

    // Async stage of the 16 x K A tile into LDS, 16 bytes per lane per issue.
    // Loop bounds are uniform across the block (16*K divisible by 2048), so
    // EXEC is all-ones for every async issue.
    for (int i8 = threadIdx.x * 8; i8 < 16 * K; i8 += blockDim.x * 8) {
        int r = i8 / K;
        int c = i8 - r * K;
        const __bf16* src;
        if (c < 512) src = A0 + (size_t)(mrow + r) * 512 + c;
        else         src = A1 + (size_t)(mrow + r) * 512 + (c - 512);
        unsigned ldsa = (unsigned)(uintptr_t)(&As[r * K + c]); // LDS byte offset
        asm volatile("global_load_async_to_lds_b128 %0, %1, off"
                     :: "v"(ldsa), "v"(src) : "memory");
    }
    asm volatile("s_wait_asynccnt 0x0" ::: "memory");
    __syncthreads();

    // A fragment addressing (16x32 bf16): lanes 0-15 M=lane, K {0..15};
    // lanes 16-31 M=lane-16, K {16..31}.
    const int m16  = lane & 15;
    const int koff = (lane >> 4) * 16;

    v8f acc = {};
    const __bf16* bptr = Bp + ((size_t)ntile * ktiles * 32 + lane) * 16;
#pragma unroll 4
    for (int kt = 0; kt < ktiles; ++kt) {
        AFrag a;
        a.h[0] = *(const v8bf*)(&As[m16 * K + kt * 32 + koff]);
        a.h[1] = *(const v8bf*)(&As[m16 * K + kt * 32 + koff + 8]);
        v16bf b = *(const v16bf*)bptr;
        __builtin_prefetch(bptr + 512, 0, 3);   // next k-tile, near-cache scope
        acc = __builtin_amdgcn_wmma_f32_16x16x32_bf16(
                  false, a.v, false, b, (short)0, acc, false, false);
        bptr += 512;                            // 32 lanes * 16 bf16
    }

    // C/D layout: element r -> M = r + (lane>=16 ? 8 : 0), N = lane & 15.
    const int cm = (lane >> 4) * 8;
    const int cn = blockIdx.y * 128 + w * 16 + (lane & 15);
    float badd = bias ? bias[cn] : 0.f;
#pragma unroll
    for (int r = 0; r < 8; ++r) {
        C[(size_t)(mrow + cm + r) * ldc + cn] = acc[r] + badd;
    }
}

// ---------------------------------------------------------------------------
// Pointwise LSTM cell. One thread per (b, ci, hidx) quaternion-state element.
// gates layout n = ci*512 + gate*128 + hidx. For layer 0, the input
// projection hqmm(x, Wx0) (in-dim == 1) is folded in here: 16 scalar MACs.
// Writes fp32 c state, bf16 h state (next-step GEMM A operand), and bf16
// h history (consumed by the next layer / attention GEMM).
// ---------------------------------------------------------------------------
__global__ void lstm_cell(const float* __restrict__ gates,
                          const float* __restrict__ x, int s,
                          const float* __restrict__ Wx0,
                          const float* __restrict__ bvec,
                          float* __restrict__ cstate,
                          __bf16* __restrict__ hstate,
                          __bf16* __restrict__ hist) {
    int tid  = blockIdx.x * blockDim.x + threadIdx.x;  // B*512 = 65536
    int b    = tid >> 9;
    int q    = tid & 511;
    int ci   = q >> 7;
    int hidx = q & 127;

    float pre[4];
#pragma unroll
    for (int gate = 0; gate < 4; ++gate) {
        int gout = gate * 128 + hidx;
        int n    = ci * 512 + gout;
        float v  = gates[(size_t)b * 2048 + n] + bvec[gout * 4 + ci];
        if (x) {
#pragma unroll
            for (int cin = 0; cin < 4; ++cin) {
                float xv = x[((size_t)b * 512 + s) * 4 + cin];
                v += xv * QSIGN[cin][ci] * Wx0[QCOMP[cin][ci] * 512 + gout];
            }
        }
        pre[gate] = v;
    }
    float ig = 1.f / (1.f + expf(-pre[0]));
    float fg = 1.f / (1.f + expf(-pre[1]));
    float gg = tanhf(pre[2]);
    float og = 1.f / (1.f + expf(-pre[3]));
    float c  = fg * cstate[tid] + ig * gg;
    cstate[tid] = c;
    float h = og * tanhf(c);
    __bf16 hb = (__bf16)h;
    hstate[tid] = hb;
    hist[(size_t)s * 65536 + tid] = hb;
}

// ---------------------------------------------------------------------------
// scores[m] = tanh(proj[m,:] @ Wa + ba) @ va,  m = s*B + b. One block per m.
// ---------------------------------------------------------------------------
__global__ void attn_scores(const float* __restrict__ proj,
                            const float* __restrict__ Wa,
                            const float* __restrict__ ba,
                            const float* __restrict__ va,
                            float* __restrict__ scores) {
    int m = blockIdx.x;
    int j = threadIdx.x;                // 128 threads
    __shared__ float prow[128];
    __shared__ float red[128];
    prow[j] = proj[(size_t)m * 128 + j];
    __syncthreads();
    float acc = ba[j];
#pragma unroll 4
    for (int p = 0; p < 128; ++p) acc += prow[p] * Wa[p * 128 + j];
    red[j] = tanhf(acc) * va[j];
    __syncthreads();
    for (int o = 64; o; o >>= 1) { if (j < o) red[j] += red[j + o]; __syncthreads(); }
    if (j == 0) scores[m] = red[0];
}

// ---------------------------------------------------------------------------
// Softmax over S per batch, weighted context over proj, then out = ctx@Wo+bo.
// One block (128 threads) per batch element.
// ---------------------------------------------------------------------------
__global__ void attn_final(const float* __restrict__ scores,
                           const float* __restrict__ proj,
                           const float* __restrict__ Wo,
                           const float* __restrict__ bo,
                           float* __restrict__ out) {
    int b = blockIdx.x;
    int t = threadIdx.x;                // 128 threads
    __shared__ float red[128];

    float mx = -1e30f;
    for (int s = t; s < 512; s += 128) mx = fmaxf(mx, scores[(size_t)s * 128 + b]);
    red[t] = mx; __syncthreads();
    for (int o = 64; o; o >>= 1) { if (t < o) red[t] = fmaxf(red[t], red[t + o]); __syncthreads(); }
    mx = red[0]; __syncthreads();

    float se = 0.f;
    for (int s = t; s < 512; s += 128) se += expf(scores[(size_t)s * 128 + b] - mx);
    red[t] = se; __syncthreads();
    for (int o = 64; o; o >>= 1) { if (t < o) red[t] += red[t + o]; __syncthreads(); }
    se = red[0]; __syncthreads();

    float ctx = 0.f;                    // thread t owns projected component t
    for (int s = 0; s < 512; ++s) {
        float a = expf(scores[(size_t)s * 128 + b] - mx);
        ctx += a * proj[((size_t)s * 128 + b) * 128 + t];
    }
    ctx /= se;
    red[t] = ctx * Wo[t]; __syncthreads();
    for (int o = 64; o; o >>= 1) { if (t < o) red[t] += red[t + o]; __syncthreads(); }
    if (t == 0) out[b] = red[0] + bo[0];
}

// ---------------------------------------------------------------------------
extern "C" void kernel_launch(void* const* d_in, const int* in_sizes, int n_in,
                              void* d_out, int out_size, void* d_ws, size_t ws_size,
                              hipStream_t stream) {
    const float* x   = (const float*)d_in[0];
    const float* Wx0 = (const float*)d_in[1];
    const float* Wh0 = (const float*)d_in[2];
    const float* b0  = (const float*)d_in[3];
    const float* Wx1 = (const float*)d_in[4];
    const float* Wh1 = (const float*)d_in[5];
    const float* b1  = (const float*)d_in[6];
    const float* Wp  = (const float*)d_in[7];
    const float* bp  = (const float*)d_in[8];
    const float* Wa  = (const float*)d_in[9];
    const float* ba  = (const float*)d_in[10];
    const float* va  = (const float*)d_in[11];
    const float* Wo  = (const float*)d_in[12];
    const float* bo  = (const float*)d_in[13];
    float* out = (float*)d_out;

    char* ws = (char*)d_ws;
    size_t off = 0;
    auto take = [&](size_t bytes) -> void* {
        void* p = (void*)(ws + off);
        off += (bytes + 255) & ~(size_t)255;
        return p;
    };
    __bf16* Wh0p   = (__bf16*)take((size_t)128 * 16 * 32 * 16 * 2);  //   2 MB
    __bf16* W1p    = (__bf16*)take((size_t)128 * 32 * 32 * 16 * 2);  //   4 MB
    __bf16* Wpp    = (__bf16*)take((size_t)  8 * 16 * 32 * 16 * 2);  // 128 KB
    __bf16* hstate = (__bf16*)take((size_t)128 * 512 * 2);
    float*  cstate = (float*) take((size_t)128 * 512 * 4);
    float*  gates  = (float*) take((size_t)128 * 2048 * 4);
    __bf16* h1hist = (__bf16*)take((size_t)512 * 128 * 512 * 2);     //  64 MB
    __bf16* h2hist = (__bf16*)take((size_t)512 * 128 * 512 * 2);     //  64 MB
    float*  proj   = (float*) take((size_t)512 * 128 * 128 * 4);     //  32 MB
    float*  scores = (float*) take((size_t)512 * 128 * 4);

    // Weight packing (Hamilton expansion -> bf16 B fragments).
    pack_hamilton<<<4096, 256, 0, stream>>>(Wh0, nullptr, 16, Wh0p);
    pack_hamilton<<<8192, 256, 0, stream>>>(Wh1, Wx1,     32, W1p);
    pack_wp<<<256, 256, 0, stream>>>(Wp, Wpp);

    // ----- Layer 0 recurrence: gates = h@Wh0big (WMMA) + fold x@Wx0 in cell.
    zero_f32<<<128, 256, 0, stream>>>((float*)hstate, 32768);
    zero_f32<<<256, 256, 0, stream>>>(cstate, 65536);
    for (int s = 0; s < 512; ++s) {
        wmma_gemm<<<dim3(8, 16), 256, 0, stream>>>(hstate, nullptr, Wh0p,
                                                   gates, 2048, 16, nullptr);
        lstm_cell<<<256, 256, 0, stream>>>(gates, x, s, Wx0, b0,
                                           cstate, hstate, h1hist);
    }

    // ----- Layer 1 recurrence: gates = [h ; h1[s]] @ [Wh1big ; Wx1big].
    zero_f32<<<128, 256, 0, stream>>>((float*)hstate, 32768);
    zero_f32<<<256, 256, 0, stream>>>(cstate, 65536);
    for (int s = 0; s < 512; ++s) {
        wmma_gemm<<<dim3(8, 16), 256, 0, stream>>>(hstate,
                                                   h1hist + (size_t)s * 65536,
                                                   W1p, gates, 2048, 32, nullptr);
        lstm_cell<<<256, 256, 0, stream>>>(gates, nullptr, s, nullptr, b1,
                                           cstate, hstate, h2hist);
    }

    // ----- Attention head.
    // projected[m, 0:128] = h2[m, :] @ Wp + bp   (M = 65536, K = 512, N = 128)
    wmma_gemm<<<dim3(4096, 1), 256, 0, stream>>>(h2hist, nullptr, Wpp,
                                                 proj, 128, 16, bp);
    attn_scores<<<65536, 128, 0, stream>>>(proj, Wa, ba, va, scores);
    attn_final<<<128, 128, 0, stream>>>(scores, proj, Wo, bo, out);
}
